// RAMAutomaton_88776974008607
// MI455X (gfx1250) — 8-bit matrix operand order repaired and checked by reading, compiled, not run on hardware
//
#include <hip/hip_runtime.h>
#include <hip/hip_bf16.h>
#include <stdint.h>


#define SEQ       128
#define BATCH     256
#define INPUT_B   256
#define STATE_B   256
#define N_IN      1024
#define N_ST      256
#define K1        512
#define K2        1280
#define OUT_MAIN  ((size_t)SEQ * BATCH * N_IN)

typedef __attribute__((ext_vector_type(8))) int v8i;

__global__ void prep_S_kernel(const int* __restrict__ conn,
                              unsigned char* __restrict__ Slo,
                              unsigned char* __restrict__ Shi,
                              int N, int K) {
    int n = blockIdx.x * blockDim.x + threadIdx.x;
    if (n >= N) return;
    unsigned* lo32 = (unsigned*)(Slo + n * K);
    unsigned* hi32 = (unsigned*)(Shi + n * K);
    for (int i = 0; i < K / 4; ++i) { lo32[i] = 0u; hi32[i] = 0u; }
    for (int k = 0; k < 16; ++k) {
        int j = conn[n * 16 + k];
        if (k < 8) Slo[n * K + j] = (unsigned char)(Slo[n * K + j] + (1u << k));
        else       Shi[n * K + j] = (unsigned char)(Shi[n * K + j] + (1u << (k - 8)));
    }
}

__device__ __forceinline__ unsigned ballot32(bool p) {
#if __has_builtin(__builtin_amdgcn_ballot_w32)
    return __builtin_amdgcn_ballot_w32(p);
#else
    return (unsigned)__ballot(p);
#endif
}

__global__ void pack_bits_kernel(const int* __restrict__ src,
                                 unsigned* __restrict__ dst) {
    unsigned gid = blockIdx.x * blockDim.x + threadIdx.x;
    bool pred = src[gid] != 0;
    unsigned m = ballot32(pred);
    if ((threadIdx.x & 31u) == 0u) dst[gid >> 5] = m;
}

__device__ __forceinline__ int opaque_zero() {
    int x = 0;
    asm volatile("" : "+v"(x));
    return x;
}

__device__ __forceinline__ v8i lds_loadA(const unsigned char* base, int stride,
                                         int lane, int c) {
    int m = lane & 15;
    int off = (lane & 16) ? 16 : 0;
    const unsigned char* p = base + m * stride + c * 64 + off;
    uint2 d0 = *(const uint2*)(p);
    uint2 d1 = *(const uint2*)(p + 8);
    uint2 d2 = *(const uint2*)(p + 32);
    uint2 d3 = *(const uint2*)(p + 40);
    v8i a;
    a[0] = (int)d0.x; a[1] = (int)d0.y; a[2] = (int)d1.x; a[3] = (int)d1.y;
    a[4] = (int)d2.x; a[5] = (int)d2.y; a[6] = (int)d3.x; a[7] = (int)d3.y;
    return a;
}

__device__ __forceinline__ v8i g_loadB(const unsigned char* __restrict__ S,
                                       int colBase, int lane) {
    int off = (lane & 16) ? 16 : 0;
    const int4* p = (const int4*)(S + colBase + off);
    int4 b0 = p[0];
    int4 b1 = p[2];
    v8i b;
    b[0] = b0.x; b[1] = b0.y; b[2] = b0.z; b[3] = b0.w;
    b[4] = b1.x; b[5] = b1.y; b[6] = b1.z; b[7] = b1.w;
    return b;
}

__global__ __launch_bounds__(512, 1) void ram_rnn_kernel(
    const int* __restrict__ in_bits,
    const int* __restrict__ init_state,
    const unsigned* __restrict__ pk_in,
    const unsigned* __restrict__ pk_st,
    const unsigned char* __restrict__ Sin_lo, const unsigned char* __restrict__ Sin_hi,
    const unsigned char* __restrict__ Sst_lo, const unsigned char* __restrict__ Sst_hi,
    float* __restrict__ out) {
    __shared__ unsigned char A1[16 * K1];
    __shared__ unsigned char A2[16 * K2];

    const int tid   = threadIdx.x;
    const int lane  = tid & 31;
    const int wave  = tid >> 5;
    const int b0    = blockIdx.x * 16;
    const int nlane = lane & 15;
    const int mbase = (lane >> 4) * 8;

#pragma unroll
    for (int r = 0; r < 8; ++r) {
        int i = tid * 8 + r;
        int m = i >> 8, n = i & 255;
        unsigned char v = (init_state[(b0 + m) * 256 + n] != 0) ? 1 : 0;
        A1[m * K1 + INPUT_B + n] = v;
        A2[m * K2 + N_IN + n]    = v;
    }

    unsigned nsbit[8];

#pragma unroll 1
    for (int t = 0; t < SEQ; ++t) {
        const int z = opaque_zero();

#pragma unroll
        for (int r = 0; r < 8; ++r) {
            int i = tid * 8 + r;
            int m = i >> 8, j = i & 255;
            A1[m * K1 + j] =
                (in_bits[((size_t)(t * BATCH + b0 + m)) * INPUT_B + j] != 0) ? 1 : 0;
        }
        __syncthreads();

#pragma unroll 1
        for (int i = 0; i < 4; ++i) {
            int ct = wave * 4 + i;
            v8i acc_lo = {};
            v8i acc_hi = {};
#pragma unroll 2
            for (int c = 0; c < K1 / 64; ++c) {
                v8i a = lds_loadA(A1, K1, lane, c);
                int colBase = (ct * 16 + nlane) * K1 + c * 64 + z;
                v8i blo = g_loadB(Sin_lo, colBase, lane);
                v8i bhi = g_loadB(Sin_hi, colBase, lane);
                acc_lo = __builtin_amdgcn_wmma_i32_16x16x64_iu8(
                    false, a, false, blo, acc_lo, false, false);
                acc_hi = __builtin_amdgcn_wmma_i32_16x16x64_iu8(
                    false, a, false, bhi, acc_hi, false, false);
            }
            int n = ct * 16 + nlane;
            const unsigned* tab = pk_in + n * 2048;
#pragma unroll
            for (int v = 0; v < 8; ++v) {
                int addr = acc_lo[v] + (acc_hi[v] << 8);
                unsigned bit = (tab[addr >> 5] >> (addr & 31)) & 1u;
                int m = mbase + v;
                out[((size_t)(t * BATCH + b0 + m)) * N_IN + n] = (float)bit;
                A2[m * K2 + n] = (unsigned char)bit;
            }
        }
        __syncthreads();

        {
            int ct = wave;
            v8i acc_lo = {};
            v8i acc_hi = {};
#pragma unroll 2
            for (int c = 0; c < K2 / 64; ++c) {
                v8i a = lds_loadA(A2, K2, lane, c);
                int colBase = (ct * 16 + nlane) * K2 + c * 64 + z;
                v8i blo = g_loadB(Sst_lo, colBase, lane);
                v8i bhi = g_loadB(Sst_hi, colBase, lane);
                acc_lo = __builtin_amdgcn_wmma_i32_16x16x64_iu8(
                    false, a, false, blo, acc_lo, false, false);
                acc_hi = __builtin_amdgcn_wmma_i32_16x16x64_iu8(
                    false, a, false, bhi, acc_hi, false, false);
            }
            int n = ct * 16 + nlane;
            const unsigned* tab = pk_st + n * 2048;
#pragma unroll
            for (int v = 0; v < 8; ++v) {
                int addr = acc_lo[v] + (acc_hi[v] << 8);
                nsbit[v] = (tab[addr >> 5] >> (addr & 31)) & 1u;
            }
        }
        __syncthreads();

        {
            int n = wave * 16 + nlane;
#pragma unroll
            for (int v = 0; v < 8; ++v) {
                int m = mbase + v;
                unsigned char b = (unsigned char)nsbit[v];
                A1[m * K1 + INPUT_B + n] = b;
                A2[m * K2 + N_IN + n]    = b;
                if (t == SEQ - 1)
                    out[OUT_MAIN + (size_t)(b0 + m) * N_ST + n] = (float)b;
            }
        }
    }
}

extern "C" void kernel_launch(void* const* d_in, const int* in_sizes, int n_in,
                              void* d_out, int out_size, void* d_ws, size_t ws_size,
                              hipStream_t stream) {
    const int* in_bits    = (const int*)d_in[0];
    const int* init_state = (const int*)d_in[1];
    const int* in_conn    = (const int*)d_in[2];
    const int* in_mem     = (const int*)d_in[3];
    const int* st_conn    = (const int*)d_in[4];
    const int* st_mem     = (const int*)d_in[5];
    float* out = (float*)d_out;

    unsigned char* ws = (unsigned char*)d_ws;
    unsigned char* Sin_lo = ws;
    unsigned char* Sin_hi = ws + 524288;
    unsigned char* Sst_lo = ws + 1048576;
    unsigned char* Sst_hi = ws + 1376256;
    unsigned*      pk_in  = (unsigned*)(ws + 1703936);
    unsigned*      pk_st  = (unsigned*)(ws + 10092544);

    prep_S_kernel<<<4, 256, 0, stream>>>(in_conn, Sin_lo, Sin_hi, N_IN, K1);
    prep_S_kernel<<<1, 256, 0, stream>>>(st_conn, Sst_lo, Sst_hi, N_ST, K2);
    pack_bits_kernel<<<(N_IN * 65536) / 256, 256, 0, stream>>>(in_mem, pk_in);
    pack_bits_kernel<<<(N_ST * 65536) / 256, 256, 0, stream>>>(st_mem, pk_st);

    ram_rnn_kernel<<<BATCH / 16, 512, 0, stream>>>(
        in_bits, init_state, pk_in, pk_st,
        Sin_lo, Sin_hi, Sst_lo, Sst_hi, out);
}
